// ChEBIRecNN_89756226551866
// MI455X (gfx1250) — matrix operand order repaired and checked by reading, not compiled, not run on hardware
//
#include <hip/hip_runtime.h>
#include <hip/hip_bf16.h>

typedef __attribute__((ext_vector_type(16))) _Float16 v16h;
typedef __attribute__((ext_vector_type(8)))  float    v8f;

#define DAGS 256
#define NODES 1024
#define L 104
#define PRED 8
#define NC 500
#define DPG 16              // DAGs per workgroup (== WMMA M tile)
#define KPAD 224            // 2L=208 padded to 7 * 32
#define JPAD 112            // L=104 padded to 7 * 16
#define KSTEPS 7            // KPAD / 32

// Load one 16x32 fp16 A fragment (ISA layout: row m = lane%16; halves 0..7 =
// K kb+base.., halves 8..15 = K kb+16+base.., base 0/8 by lane half).
#define LOADA(dst, ks)                                                       \
    do {                                                                     \
        const int a0_ = (ks) * 32 + ((lane < 16) ? 0 : 8);                   \
        _Pragma("unroll")                                                    \
        for (int h_ = 0; h_ < 8; ++h_) (dst)[h_]     = sX[m_][a0_ + h_];     \
        _Pragma("unroll")                                                    \
        for (int h_ = 0; h_ < 8; ++h_) (dst)[8 + h_] = sX[m_][a0_ + 16 + h_];\
    } while (0)

#define WMMA(a_, ks)                                                         \
    accv = __builtin_amdgcn_wmma_f32_16x16x32_f16(                           \
        false, (a_), false, bfrag[ks], (short)0, accv, false, false)

// ---------------------------------------------------------------------------
// Kernel 1: one workgroup scans 16 DAGs in lockstep through the 1024-node
// topological order. Per node: gather predecessor outputs from global history
// (fp32, L2 resident), per-feature softmax-attention aggregate -> fp16 X tile
// in LDS, then a [16x208]x[208x104] GEMM via v_wmma_f32_16x16x32_f16.
// Loop-invariant: 7 B fragments (W_merge^T) + bias in VGPRs; history stores
// use a running integer offset so they stay on the GLOBAL path with constant
// per-row immediate offsets; A fragments are double-buffered so LDS loads of
// step k+1 overlap the WMMA of step k.
// ---------------------------------------------------------------------------
__global__ void __launch_bounds__(256)
dag_scan_kernel(const float* __restrict__ atom,      // [D,N,L]
                const int*   __restrict__ pred,      // [D,N,P]
                const float* __restrict__ W_single,  // [L,L]
                const float* __restrict__ b_single,  // [L]
                const float* __restrict__ W_merge,   // [L,2L]
                const float* __restrict__ b_merge,   // [L]
                const float* __restrict__ attn_w,    // [L]
                float* __restrict__ hist)            // [D,N,L] recurrent state
{
    __shared__ _Float16 sB[KPAD][JPAD];   // W_merge^T fp16, staged once (49KB)
    __shared__ _Float16 sX[DPG][KPAD];    // per-step activations (7KB)
    __shared__ float    s_attn[L];
    __shared__ float    s_bm[L];
    __shared__ int      s_pred[DPG][PRED];

    const int tid  = threadIdx.x;
    const int g    = blockIdx.x;          // group of 16 DAGs
    const int wave = tid >> 5;
    const int lane = tid & 31;

    // ---- one-time staging: W_merge^T -> LDS fp16, biases, attn weights ----
    for (int it = tid; it < KPAD * JPAD; it += 256) {
        const int k = it / JPAD, j = it % JPAD;
        _Float16 v = (_Float16)0.0f;
        if (k < 2 * L && j < L) v = (_Float16)W_merge[j * (2 * L) + k];
        sB[k][j] = v;
    }
    if (tid < L) { s_attn[tid] = attn_w[tid]; s_bm[tid] = b_merge[tid]; }
    // zero the K pad of X once (208..223 never rewritten)
    for (int it = tid; it < DPG * (KPAD - 2 * L); it += 256) {
        sX[it / (KPAD - 2 * L)][2 * L + it % (KPAD - 2 * L)] = (_Float16)0.0f;
    }

    // ---- node 0: root path relu(W_single @ feat + b_single), VALU matvec ----
    for (int it = tid; it < DPG * L; it += 256) {
        const int m = it / L, j = it % L;
        const size_t dag = (size_t)(g * DPG + m);
        const float* feat = atom + (dag * NODES + 0) * L;
        float acc = b_single[j];
        const float* wrow = W_single + (size_t)j * L;
        #pragma unroll 8
        for (int l = 0; l < L; ++l) acc += wrow[l] * feat[l];
        hist[(dag * NODES + 0) * L + j] = fmaxf(acc, 0.0f);
    }
    __syncthreads();

    // ---- hoist loop-invariant state (waves 0..6) ----
    // B layout per ISA: col j = j0 + lane%16; halves 0..7 = K kb+base..,
    // halves 8..15 = K kb+16+base.., base 0/8 by lane half (matches A).
    v16h bfrag[KSTEPS];
    float bias = 0.0f;
    const int col = wave * 16 + (lane & 15);
    const bool colok = (wave < 7) && (col < L);
    // running element offset into hist for C row r=0 of this lane; rows are
    // spaced by the compile-time constant NODES*L elements (fits IOFFSET).
    size_t obase = 0;
    if (wave < 7) {
        #pragma unroll
        for (int ks = 0; ks < KSTEPS; ++ks) {
            const int k0 = ks * 32 + ((lane < 16) ? 0 : 8);
            #pragma unroll
            for (int h = 0; h < 16; ++h) bfrag[ks][h] = sB[k0 + (h < 8 ? h : h + 8)][col];
        }
        if (colok) {
            bias = s_bm[col];
            const int row0 = (lane < 16) ? 0 : 8;
            obase = ((size_t)(g * DPG + row0) * NODES + 1) * L + col;
        }
    }

    // ---- sequential scan over nodes 1..N-1 ----
    for (int n = 1; n < NODES; ++n) {
        // stage this node's predecessor indices for the 16 DAGs
        if (tid < DPG * PRED) {
            const int m = tid >> 3, p = tid & 7;
            s_pred[m][p] =
                pred[((size_t)(g * DPG + m) * NODES + n) * PRED + p];
        }
        __syncthreads();

        // gather + per-feature softmax over predecessors -> X (fp16 in LDS)
        for (int it = tid; it < DPG * L; it += 256) {
            const int m = it / L, l = it % L;
            const size_t dag = (size_t)(g * DPG + m);
            const float aw = s_attn[l];
            float po[PRED];
            int   pv[PRED];
            float mx = -3.0e38f;
            #pragma unroll
            for (int p = 0; p < PRED; ++p) {
                const int pi = s_pred[m][p];
                if (pi >= 0) {
                    const float v = hist[(dag * NODES + pi) * L + l];
                    po[p] = v; pv[p] = 1;
                    mx = fmaxf(mx, aw * v);
                } else { po[p] = 0.0f; pv[p] = 0; }
            }
            float ssum = 0.0f, acc = 0.0f;
            #pragma unroll
            for (int p = 0; p < PRED; ++p) {
                if (pv[p]) {
                    const float e = __expf(aw * po[p] - mx);
                    ssum += e; acc += e * po[p];
                }
            }
            const float agg = (ssum > 0.0f) ? (acc / ssum) : 0.0f;
            sX[m][l]     = (_Float16)agg;
            sX[m][L + l] = (_Float16)atom[(dag * NODES + n) * L + l];
        }
        __syncthreads();

        if (wave < 7) {
            // double-buffered A fragments: loads for step k+1 are issued
            // before the WMMA of step k so the XDL pipe overlaps LDS latency.
            const int m_ = lane & 15;
            v16h a0, a1;
            v8f accv = {};
            LOADA(a0, 0);
            LOADA(a1, 1);
            WMMA(a0, 0);
            LOADA(a0, 2);
            WMMA(a1, 1);
            LOADA(a1, 3);
            WMMA(a0, 2);
            LOADA(a0, 4);
            WMMA(a1, 3);
            LOADA(a1, 5);
            WMMA(a0, 4);
            LOADA(a0, 6);
            WMMA(a1, 5);
            WMMA(a0, 6);
            // C layout: VGPR r -> row (lane<16 ? r : 8+r), col = lane%16
            if (colok) {
                #pragma unroll
                for (int r = 0; r < 8; ++r) {
                    const float v = accv[r] + bias;
                    hist[obase + (size_t)r * ((size_t)NODES * L)] =
                        fmaxf(v, 0.0f);
                }
                obase += L;      // advance to next node's slot
            }
        } else {
            // wave 7: prefetch next node's features (global_prefetch_b8)
            if (lane < DPG && n + 1 < NODES) {
                const size_t dag = (size_t)(g * DPG + lane);
                __builtin_prefetch(&atom[(dag * NODES + (n + 1)) * L], 0, 3);
            }
        }
        __syncthreads();   // hist writes visible to next iteration's gather
    }
}

// ---------------------------------------------------------------------------
// Kernel 2: softmax-pool over the DAG dimension (reading the sink-node row
// of hist directly), then 500x104 classifier.
// ---------------------------------------------------------------------------
__global__ void __launch_bounds__(256)
finalize_kernel(const float* __restrict__ hist,     // [D,N,L]
                const float* __restrict__ dag_w,    // [L]
                const float* __restrict__ W_final,  // [NC,L]
                const float* __restrict__ b_final,  // [NC]
                float* __restrict__ out)            // [NC]
{
    __shared__ float pooled[L];
    const int tid = threadIdx.x;
    if (tid < L) {
        const float w = dag_w[tid];
        float mx = -3.0e38f;
        for (int d = 0; d < DAGS; ++d)
            mx = fmaxf(mx, w * hist[((size_t)d * NODES + NODES - 1) * L + tid]);
        float s = 0.0f, a = 0.0f;
        for (int d = 0; d < DAGS; ++d) {
            const float v = hist[((size_t)d * NODES + NODES - 1) * L + tid];
            const float e = __expf(w * v - mx);
            s += e; a += e * v;
        }
        pooled[tid] = a / s;
    }
    __syncthreads();
    for (int c = tid; c < NC; c += 256) {
        float acc = b_final[c];
        const float* wrow = W_final + (size_t)c * L;
        #pragma unroll 8
        for (int l = 0; l < L; ++l) acc += wrow[l] * pooled[l];
        out[c] = acc;
    }
}

extern "C" void kernel_launch(void* const* d_in, const int* in_sizes, int n_in,
                              void* d_out, int out_size, void* d_ws, size_t ws_size,
                              hipStream_t stream) {
    const float* atom     = (const float*)d_in[0];
    const int*   pred     = (const int*)  d_in[1];
    const float* W_single = (const float*)d_in[2];
    const float* b_single = (const float*)d_in[3];
    const float* W_merge  = (const float*)d_in[4];
    const float* b_merge  = (const float*)d_in[5];
    const float* attn_w   = (const float*)d_in[6];
    const float* dag_w    = (const float*)d_in[7];
    const float* W_final  = (const float*)d_in[8];
    const float* b_final  = (const float*)d_in[9];
    float* outp = (float*)d_out;

    float* hist = (float*)d_ws;          // D*N*L fp32 (~109 MB, L2 resident)

    dag_scan_kernel<<<DAGS / DPG, 256, 0, stream>>>(
        atom, pred, W_single, b_single, W_merge, b_merge, attn_w, hist);
    finalize_kernel<<<1, 256, 0, stream>>>(
        hist, dag_w, W_final, b_final, outp);
}
